// LinearMultiheadAttention_69913477644662
// MI455X (gfx1250) — compile-verified, matrix operand order checked by reading, not run on hardware
//
#include <hip/hip_runtime.h>
#include <hip/hip_bf16.h>

// Linear multi-head attention (elu+1 feature map), MI455X / gfx1250.
// Matrix products: v_wmma_f32_16x16x32_f16. B-operand staging for the two
// large GEMMs uses the CDNA5 async global->LDS engine (ASYNCcnt).

typedef _Float16 half_t;
typedef __attribute__((ext_vector_type(16))) _Float16 v16h;
typedef __attribute__((ext_vector_type(8)))  _Float16 v8h;
typedef __attribute__((ext_vector_type(8)))  float    v8f;

#define NB    4
#define NH    12
#define NS    8192
#define CDIM  768
#define DH    64
#define MTOT  (NB * NS)    // 32768
#define BHTOT (NB * NH)    // 48

// Fragment layout convention (self-consistent across A/B operands):
//   lane L in [0,16): holds row/col (L) of the tile, halves = K .. K+15
//   lane L in [16,32): same row/col (L-16), halves = K+16 .. K+31
// => each fragment is one contiguous 32-byte load per lane.
__device__ __forceinline__ v16h load_frag(const half_t* p) {
  v8h a = *(const v8h*)(p);
  v8h b = *(const v8h*)(p + 8);
  v16h r;
#pragma unroll
  for (int i = 0; i < 8; ++i) { r[i] = a[i]; r[i + 8] = b[i]; }
  return r;
}

__device__ __forceinline__ v8f wmma_f16(v16h a, v16h b, v8f c) {
  return __builtin_amdgcn_wmma_f32_16x16x32_f16(
      false, a, false, b, (short)0, c, false, false);
}

__device__ __forceinline__ float phi(float x) {  // elu(x)+1
  return x > 0.f ? x + 1.f : __expf(x);
}

// Async copy 16B global -> LDS (CDNA5 async engine, tracked by ASYNCcnt).
// lds_off = low 32 bits of the generic pointer (= LDS byte address).
__device__ __forceinline__ void async_g2l_b128(uint32_t lds_off, const half_t* g) {
  asm volatile("global_load_async_to_lds_b128 %0, %1, off"
               :: "v"(lds_off), "v"(g) : "memory");
}

// ---------------------------------------------------------------- conversions
__global__ void cvt_x_kernel(const float* __restrict__ x, half_t* __restrict__ xh, int n) {
  int i = blockIdx.x * blockDim.x + threadIdx.x;
  if (i < n) xh[i] = (half_t)x[i];
}

// Transpose+convert weights: WqT[c][k] = W_qkv[k][c], WoT[c][k] = W_out[k][c]
__global__ void prep_w_kernel(const float* __restrict__ Wqkv, const float* __restrict__ Wout,
                              half_t* __restrict__ WqT, half_t* __restrict__ WoT) {
  int i = blockIdx.x * blockDim.x + threadIdx.x;
  if (i < 3 * CDIM * CDIM) {
    int c = i / CDIM, k = i % CDIM;
    WqT[i] = (half_t)Wqkv[(size_t)k * (3 * CDIM) + c];
  }
  if (i < CDIM * CDIM) {
    int c = i / CDIM, k = i % CDIM;
    WoT[i] = (half_t)Wout[(size_t)k * CDIM + c];
  }
}

// ------------------------------------------------- QKV GEMM + phi + reshapes
// [MTOT x 768] @ [768 x 2304].  Block tile = 128 rows x 128 cols, 8 waves
// (wave = 16 rows x 128 cols = 8 WMMA tiles).  B tiles (32k x 128c) are
// staged in LDS by the async global->LDS engine, double-buffered.
__global__ __launch_bounds__(256)
void qkv_gemm_kernel(const half_t* __restrict__ xh, const half_t* __restrict__ WqT,
                     const float* __restrict__ bqkv,
                     half_t* __restrict__ qf,    // [BH][N][D]
                     half_t* __restrict__ kfT,   // [BH][D][N]
                     half_t* __restrict__ vT) {  // [BH][D][N]
  __shared__ __align__(16) half_t lds_b[2][128 * 32];  // [stage][col][k] 16 KiB

  const int wave = threadIdx.x >> 5;
  const int lane = threadIdx.x & 31;
  const int lo = lane & 15, hi = lane >> 4;
  const int m0 = blockIdx.y * 128 + wave * 16;
  const int c0 = blockIdx.x * 128;

  // Each thread copies 2 x 16B per stage: 512 chunks cover 128 cols x 32 k.
  auto issue_stage = [&](int stage, int k0) {
#pragma unroll
    for (int u = 0; u < 2; ++u) {
      const int chunk = threadIdx.x * 2 + u;   // 0..511
      const int c = chunk >> 2;                // column within tile
      const int part = chunk & 3;              // 8-half slice of the 32 k's
      const uint32_t loff =
          (uint32_t)(uintptr_t)&lds_b[stage][c * 32 + part * 8];
      async_g2l_b128(loff, WqT + (size_t)(c0 + c) * CDIM + k0 + part * 8);
    }
  };

  v8f acc[8] = {};
  const half_t* arow = xh + (size_t)(m0 + lo) * CDIM + hi * 16;

  issue_stage(0, 0);
  const int KSTEPS = CDIM / 32;  // 24
#pragma unroll 1
  for (int kc = 0; kc < KSTEPS; ++kc) {
    const int cur = kc & 1;
    if (kc + 1 < KSTEPS) {
      issue_stage(cur ^ 1, (kc + 1) * 32);
      asm volatile("s_wait_asynccnt 0x2" ::: "memory");  // current stage done
    } else {
      asm volatile("s_wait_asynccnt 0x0" ::: "memory");
    }
    __syncthreads();
    v16h a = load_frag(arow + kc * 32);
#pragma unroll
    for (int j = 0; j < 8; ++j) {
      v16h b = load_frag(&lds_b[cur][(16 * j + lo) * 32 + hi * 16]);
      acc[j] = wmma_f16(a, b, acc[j]);
    }
    __syncthreads();  // all waves done reading `cur` before it is overwritten
  }

#pragma unroll
  for (int j = 0; j < 8; ++j) {
    const int ct = c0 + 16 * j;          // tile base column
    const int s  = ct / CDIM;            // 0=q 1=k 2=v (constant per tile)
    const int h  = (ct % CDIM) / DH;     // head (constant per tile)
    const int d  = (ct % DH) + lo;
    const float bias = bqkv[ct + lo];
#pragma unroll
    for (int i = 0; i < 8; ++i) {
      const int m  = m0 + i + 8 * hi;    // C-tile row (ISA C/D VGPR layout)
      const int b_ = m >> 13;
      const int n  = m & (NS - 1);
      const int bh = b_ * NH + h;
      const float val = acc[j][i] + bias;
      if (s == 0) {
        qf[((size_t)bh * NS + n) * DH + d] = (half_t)phi(val);
      } else if (s == 1) {
        kfT[((size_t)bh * DH + d) * NS + n] = (half_t)phi(val);
      } else {
        vT[((size_t)bh * DH + d) * NS + n] = (half_t)val;
      }
    }
  }
}

// ------------------------------------------- kv[b,h] = kf^T @ v ; z = sum(kf)
__global__ __launch_bounds__(256)
void kv_agg_kernel(const half_t* __restrict__ kfT, const half_t* __restrict__ vT,
                   half_t* __restrict__ kvT,   // [BH][e=64][d=64]
                   float* __restrict__ z) {    // [BH][64]
  __shared__ float zs[DH];
  const int bh   = blockIdx.x;
  const int wave = threadIdx.x >> 5;
  const int lane = threadIdx.x & 31;
  const int lo = lane & 15, hi = lane >> 4;
  const int ti   = wave & 3;   // tile row: d in [16*ti, 16*ti+16)
  const int jset = wave >> 2;  // 0 -> e tiles {0,1}, 1 -> {2,3}

  if (threadIdx.x < DH) zs[threadIdx.x] = 0.f;
  __syncthreads();

  const half_t* kh = kfT + (size_t)bh * DH * NS;
  const half_t* vh = vT  + (size_t)bh * DH * NS;

  v8f acc[2] = {};
  float zp = 0.f;
#pragma unroll 1
  for (int n0 = 0; n0 < NS; n0 += 32) {
    v16h a = load_frag(kh + (size_t)(16 * ti + lo) * NS + n0 + hi * 16);
    if (jset == 0) {
#pragma unroll
      for (int t = 0; t < 16; ++t) zp += (float)a[t];
    }
#pragma unroll
    for (int jj = 0; jj < 2; ++jj) {
      const int e0 = (jset * 2 + jj) * 16;
      v16h b = load_frag(vh + (size_t)(e0 + lo) * NS + n0 + hi * 16);
      acc[jj] = wmma_f16(a, b, acc[jj]);
    }
  }

  if (jset == 0) atomicAdd(&zs[16 * ti + lo], zp);

#pragma unroll
  for (int jj = 0; jj < 2; ++jj) {
    const int e = (jset * 2 + jj) * 16 + lo;
#pragma unroll
    for (int i = 0; i < 8; ++i) {
      const int d = 16 * ti + i + 8 * hi;
      kvT[(size_t)bh * DH * DH + (size_t)e * DH + d] = (half_t)acc[jj][i];
    }
  }
  __syncthreads();
  if (threadIdx.x < DH) z[bh * DH + threadIdx.x] = zs[threadIdx.x];
}

// ------------------------------- y = (qf @ kv) / max(qf . z, eps) -> f16 [B,N,C]
__global__ __launch_bounds__(256)
void attn_apply_kernel(const half_t* __restrict__ qf, const half_t* __restrict__ kvT,
                       const float* __restrict__ z, half_t* __restrict__ yh) {
  __shared__ float zsm[DH];
  __shared__ float sden[8][16];
  const int bh   = blockIdx.x;
  const int b_   = bh / NH, h = bh % NH;
  const int wave = threadIdx.x >> 5;
  const int lane = threadIdx.x & 31;
  const int lo = lane & 15, hi = lane >> 4;
  const int n0 = blockIdx.y * 128 + wave * 16;

  if (threadIdx.x < DH) zsm[threadIdx.x] = z[bh * DH + threadIdx.x];
  __syncthreads();

  const half_t* qrow = qf + ((size_t)bh * NS + n0 + lo) * DH + hi * 16;
  const half_t* kvb  = kvT + (size_t)bh * DH * DH;

  v8f acc[4] = {};
  float den = 0.f;
#pragma unroll
  for (int kc = 0; kc < 2; ++kc) {
    const int k0 = kc * 32;
    v16h a = load_frag(qrow + k0);
#pragma unroll
    for (int t = 0; t < 16; ++t) den += (float)a[t] * zsm[k0 + hi * 16 + t];
#pragma unroll
    for (int j = 0; j < 4; ++j) {
      v16h bfr = load_frag(kvb + (size_t)(16 * j + lo) * DH + k0 + hi * 16);
      acc[j] = wmma_f16(a, bfr, acc[j]);
    }
  }
  den += __shfl_xor(den, 16);
  if (hi == 0) sden[wave][lo] = den;
  __syncthreads();

#pragma unroll
  for (int j = 0; j < 4; ++j) {
    const int e = 16 * j + lo;
#pragma unroll
    for (int i = 0; i < 8; ++i) {
      const int r  = i + 8 * hi;
      const float dn = fmaxf(sden[wave][r], 1e-6f);
      const int n  = n0 + r;
      yh[((size_t)b_ * NS + n) * CDIM + h * DH + e] = (half_t)(acc[j][i] / dn);
    }
  }
}

// ------------------------------------------------- out = y @ W_out + b_out (f32)
// Same async-staged 128x128 block-tile scheme as the QKV GEMM.
__global__ __launch_bounds__(256)
void out_gemm_kernel(const half_t* __restrict__ yh, const half_t* __restrict__ WoT,
                     const float* __restrict__ bout, float* __restrict__ out) {
  __shared__ __align__(16) half_t lds_b[2][128 * 32];

  const int wave = threadIdx.x >> 5;
  const int lane = threadIdx.x & 31;
  const int lo = lane & 15, hi = lane >> 4;
  const int m0 = blockIdx.y * 128 + wave * 16;
  const int c0 = blockIdx.x * 128;

  auto issue_stage = [&](int stage, int k0) {
#pragma unroll
    for (int u = 0; u < 2; ++u) {
      const int chunk = threadIdx.x * 2 + u;
      const int c = chunk >> 2;
      const int part = chunk & 3;
      const uint32_t loff =
          (uint32_t)(uintptr_t)&lds_b[stage][c * 32 + part * 8];
      async_g2l_b128(loff, WoT + (size_t)(c0 + c) * CDIM + k0 + part * 8);
    }
  };

  v8f acc[8] = {};
  const half_t* arow = yh + (size_t)(m0 + lo) * CDIM + hi * 16;

  issue_stage(0, 0);
  const int KSTEPS = CDIM / 32;
#pragma unroll 1
  for (int kc = 0; kc < KSTEPS; ++kc) {
    const int cur = kc & 1;
    if (kc + 1 < KSTEPS) {
      issue_stage(cur ^ 1, (kc + 1) * 32);
      asm volatile("s_wait_asynccnt 0x2" ::: "memory");
    } else {
      asm volatile("s_wait_asynccnt 0x0" ::: "memory");
    }
    __syncthreads();
    v16h a = load_frag(arow + kc * 32);
#pragma unroll
    for (int j = 0; j < 8; ++j) {
      v16h b = load_frag(&lds_b[cur][(16 * j + lo) * 32 + hi * 16]);
      acc[j] = wmma_f16(a, b, acc[j]);
    }
    __syncthreads();
  }

#pragma unroll
  for (int j = 0; j < 8; ++j) {
    const int c = c0 + 16 * j + lo;
    const float bias = bout[c];
#pragma unroll
    for (int i = 0; i < 8; ++i) {
      const int m = m0 + i + 8 * hi;
      out[(size_t)m * CDIM + c] = acc[j][i] + bias;
    }
  }
}

// ---------------------------------------------------------------------- launch
extern "C" void kernel_launch(void* const* d_in, const int* in_sizes, int n_in,
                              void* d_out, int out_size, void* d_ws, size_t ws_size,
                              hipStream_t stream) {
  const float* x    = (const float*)d_in[0];
  const float* Wqkv = (const float*)d_in[1];
  const float* bqkv = (const float*)d_in[2];
  const float* Wout = (const float*)d_in[3];
  const float* bout = (const float*)d_in[4];
  float* out = (float*)d_out;

  size_t off = 0;
  auto carve = [&](size_t bytes) {
    void* p = (char*)d_ws + off;
    off += (bytes + 255) & ~(size_t)255;
    return p;
  };
  half_t* xh  = (half_t*)carve((size_t)MTOT * CDIM * sizeof(half_t));       // 48 MiB
  half_t* qf  = (half_t*)carve((size_t)BHTOT * NS * DH * sizeof(half_t));   // 48 MiB
  half_t* kfT = (half_t*)carve((size_t)BHTOT * DH * NS * sizeof(half_t));   // 48 MiB
  half_t* vT  = (half_t*)carve((size_t)BHTOT * DH * NS * sizeof(half_t));   // 48 MiB
  half_t* WqT = (half_t*)carve((size_t)3 * CDIM * CDIM * sizeof(half_t));   // 3.4 MiB
  half_t* WoT = (half_t*)carve((size_t)CDIM * CDIM * sizeof(half_t));       // 1.1 MiB
  half_t* kvT = (half_t*)carve((size_t)BHTOT * DH * DH * sizeof(half_t));   // 384 KiB
  float*  z   = (float*)carve((size_t)BHTOT * DH * sizeof(float));          // 12 KiB
  half_t* yh  = xh;  // stream-ordered reuse

  (void)in_sizes; (void)n_in; (void)out_size; (void)ws_size;

  const int nx = MTOT * CDIM;
  cvt_x_kernel<<<(nx + 255) / 256, 256, 0, stream>>>(x, xh, nx);
  prep_w_kernel<<<(3 * CDIM * CDIM + 255) / 256, 256, 0, stream>>>(Wqkv, Wout, WqT, WoT);

  qkv_gemm_kernel<<<dim3(3 * CDIM / 128, MTOT / 128), 256, 0, stream>>>(
      xh, WqT, bqkv, qf, kfT, vT);

  kv_agg_kernel<<<BHTOT, 256, 0, stream>>>(kfT, vT, kvT, z);

  attn_apply_kernel<<<dim3(BHTOT, NS / 128), 256, 0, stream>>>(qf, kvT, z, yh);

  out_gemm_kernel<<<dim3(CDIM / 128, MTOT / 128), 256, 0, stream>>>(yh, WoT, bout, out);
}
